// AttentionHelper_29746943492615
// MI455X (gfx1250) — matrix operand order repaired and checked by reading, not compiled, not run on hardware
//
#include <hip/hip_runtime.h>

// Problem constants (from reference setup_inputs)
#define B_     8
#define C_     256
#define L_     2048
#define BAND   (L_/2 + 64)   // 1088: keep keys m <= l + 1088
#define ROWS   16            // l-rows per workgroup
#define MCHUNK 128           // m columns staged per iteration
#define NWAVE  8
#define NTHR   256

typedef __attribute__((ext_vector_type(16))) __bf16         v16bf;
typedef __attribute__((ext_vector_type(16))) unsigned short v16us;
typedef __attribute__((ext_vector_type(8)))  float          v8f;

union Frag { v16us u; v16bf b; uint4 q[2]; unsigned int d[8]; };

// Split f32 into bf16 hi (truncated) + bf16 lo (RNE residual).
// Returned u32: low 16 = hi bits, high 16 = lo bits.
__device__ __forceinline__ unsigned int split_bf16(float x) {
  unsigned int xb = __float_as_uint(x);
  unsigned int hi = xb & 0xFFFF0000u;
  float lof = x - __uint_as_float(hi);
  unsigned int lb = __float_as_uint(lof);
  unsigned int lo = (lb + 0x7FFFu + ((lb >> 16) & 1u)) >> 16;
  return (hi >> 16) | (lo << 16);
}

// Fragment = 32 contiguous, 32B-aligned bytes in LDS -> two ds_load_b128.
__device__ __forceinline__ void ld_frag(Frag& f, const uint4* p) {
  f.q[0] = p[0];
  f.q[1] = p[1];
}

#define WMMA_BF16(A, Bm, Cm) \
  __builtin_amdgcn_wmma_f32_16x16x32_bf16(false, (A), false, (Bm), (short)0, (Cm), false, false)

extern "C" __global__ __launch_bounds__(NTHR, 1)
void attn_fused_kernel(const float* __restrict__ Q, const float* __restrict__ K,
                       const float* __restrict__ V, const float* __restrict__ pad,
                       float* __restrict__ out, float* __restrict__ attn)
{
  // LDS map (272 KB total, needs CDNA5's 320KB/WGP):
  //   [0KB   ) qH : 8 chunks * 16 rows * 32 bf16 A-frag plane (hi)   = 8KB
  //   [8KB   ) qL : same (lo)                                        = 8KB
  //   [16KB  ) S  : 16 x 2048 f32 score strip                        = 128KB
  //   [144KB ) kH : 8 chunks * 128 cols * 32 bf16 B-frag plane (hi)  = 64KB
  //   [208KB ) kL : same (lo)                                        = 64KB
  //   kH/kL are reused as attention B-frag planes (64 chunks*16*32) in phase 2.
  extern __shared__ char smem[];
  unsigned short* qH = (unsigned short*)(smem);
  unsigned short* qL = (unsigned short*)(smem + 8*1024);
  float*          S  = (float*)(smem + 16*1024);
  unsigned short* kH = (unsigned short*)(smem + 144*1024);
  unsigned short* kL = (unsigned short*)(smem + 208*1024);
  unsigned int*   kH32 = (unsigned int*)kH;
  unsigned int*   kL32 = (unsigned int*)kL;
  const uint4*    qH4 = (const uint4*)qH;
  const uint4*    qL4 = (const uint4*)qL;
  const uint4*    kH4 = (const uint4*)kH;
  const uint4*    kL4 = (const uint4*)kL;
  unsigned short* afH = kH;            // phase-2 attention planes (aliases)
  unsigned short* afL = kL;
  const uint4*    afH4 = (const uint4*)afH;
  const uint4*    afL4 = (const uint4*)afL;

  const int t    = threadIdx.x;
  const int lane = t & 31;
  const int wave = t >> 5;
  const int half = lane >> 4;   // wave32 half
  const int l16  = lane & 15;

  const int b     = blockIdx.x >> 7;     // 8 batches
  const int lt    = blockIdx.x & 127;    // 128 l-tiles of 16
  const int lBase = lt * ROWS;

  const float* Qb = Q   + (size_t)b * C_ * L_;
  const float* Kb = K   + (size_t)b * C_ * L_;
  const float* Vb = V   + (size_t)b * C_ * L_;
  const float* Pb = pad + (size_t)b * L_ * L_;
  float*       Ob = out  + (size_t)b * C_ * L_;
  float*       Ab = attn + (size_t)b * L_ * L_;

  // ---- Stage Q tile into A-fragment-major split-bf16 planes ----
  // A-frag element j of lane(mr,half) holds k = half*16 + j.
  for (int idx = t; idx < C_ * ROWS; idx += NTHR) {
    int c = idx >> 4, lr = idx & 15;
    unsigned int p = split_bf16(Qb[(size_t)c * L_ + lBase + lr]);
    int kl  = c & 31;
    int off = (((c >> 5) * ROWS + lr) << 5) + kl;
    qH[off] = (unsigned short)(p & 0xFFFFu);
    qL[off] = (unsigned short)(p >> 16);
  }
  __syncthreads();

  // ---- Phase 1: S[l][m] = sum_c Q[c][l]*K[c][m] (split-bf16, 3 WMMA) ----
  for (int chunk = 0; chunk < L_ / MCHUNK; ++chunk) {
    const int m0 = chunk * MCHUNK;
    // Stage K chunk transposed into B-frag planes [kchunk][m][klocal].
    // Two consecutive c per thread -> paired b32 LDS stores; global reads
    // stay coalesced along m.
    for (int idx = t; idx < (C_ / 2) * MCHUNK; idx += NTHR) {
      int cp = idx >> 7;             // c-pair index 0..127
      int m  = idx & (MCHUNK - 1);
      int c0 = cp << 1;
      const float* kc = Kb + (size_t)c0 * L_ + m0 + m;
      unsigned int p0 = split_bf16(kc[0]);
      unsigned int p1 = split_bf16(kc[L_]);
      int u32i = ((((c0 >> 5) * MCHUNK + m) << 5) + (c0 & 31)) >> 1;
      kH32[u32i] = (p0 & 0xFFFFu) | ((p1 & 0xFFFFu) << 16);
      kL32[u32i] = (p0 >> 16)     | (p1 & 0xFFFF0000u);
    }
    // Prefetch next K chunk -> global_prefetch_b8
    if (chunk + 1 < L_ / MCHUNK)
      __builtin_prefetch(Kb + (size_t)t * L_ + m0 + MCHUNK, 0, 3);
    __syncthreads();

    // Each wave owns a 16-wide m tile: 8 chunks of K, 3 WMMA each.
    v8f acc = {0.f,0.f,0.f,0.f,0.f,0.f,0.f,0.f};
    for (int kbc = 0; kbc < C_ / 32; ++kbc) {
      Frag ahi, alo, bhi, blo;
      const int ai = (kbc * ROWS + l16) * 4 + half * 2;                 // uint4 units
      const int bi = (kbc * MCHUNK + wave * 16 + l16) * 4 + half * 2;
      ld_frag(ahi, qH4 + ai); ld_frag(alo, qL4 + ai);
      ld_frag(bhi, kH4 + bi); ld_frag(blo, kL4 + bi);
      acc = WMMA_BF16(ahi.b, bhi.b, acc);   // hi*hi
      acc = WMMA_BF16(ahi.b, blo.b, acc);   // hi*lo
      acc = WMMA_BF16(alo.b, bhi.b, acc);   // lo*hi
    }
#pragma unroll
    for (int r = 0; r < 8; ++r) {
      int M = r + 8 * half;                 // D: M = r + 8*half, N = lane%16
      S[M * L_ + m0 + wave * 16 + l16] = acc[r];
    }
    __syncthreads();
  }

  // ---- Softmax rows: s = e*mask/16 + log(mask+1e-6); normalize ----
  const float scale = 0.0625f;  // 1/sqrt(256)
  for (int row = wave; row < ROWS; row += NWAVE) {
    const int l = lBase + row;
    const float* prow = Pb + (size_t)l * L_;
    float mx = -3.402823e38f;
    for (int i = lane; i < L_; i += 32) {
      float mv = (i <= l + BAND) ? prow[i] : 0.f;
      float s  = S[row * L_ + i] * mv * scale + __logf(mv + 1e-6f);
      S[row * L_ + i] = s;
      mx = fmaxf(mx, s);
    }
#pragma unroll
    for (int o = 16; o > 0; o >>= 1) mx = fmaxf(mx, __shfl_xor(mx, o, 32));
    float sum = 0.f;
    for (int i = lane; i < L_; i += 32) {
      float e = __expf(S[row * L_ + i] - mx);
      S[row * L_ + i] = e;
      sum += e;
    }
#pragma unroll
    for (int o = 16; o > 0; o >>= 1) sum += __shfl_xor(sum, o, 32);
    const float inv = 1.f / sum;
    float* arow = Ab + (size_t)l * L_;
    for (int i = lane; i < L_; i += 32) {
      float a = S[row * L_ + i] * inv;
      arow[i] = a;                          // attention output (coalesced)
      unsigned int p = split_bf16(a);       // B-frag planes for phase 2
      int off = (((i >> 5) * ROWS + row) << 5) + (i & 31);
      afH[off] = (unsigned short)(p & 0xFFFFu);
      afL[off] = (unsigned short)(p >> 16);
    }
  }
  __syncthreads();

  // ---- Phase 2: out[c][l] = sum_m V[c][m] * A[l][m] ----
  for (int ct = wave; ct < C_ / 16; ct += NWAVE) {
    const int cBase = ct * 16;
    const float* vrow = Vb + (size_t)(cBase + l16) * L_;  // A-matrix row = c
    v8f acc = {0.f,0.f,0.f,0.f,0.f,0.f,0.f,0.f};
    for (int kbc = 0; kbc < L_ / 32; ++kbc) {
      const int kb = kbc << 5;
      float va[16];
      *(float4*)&va[0]  = *(const float4*)(vrow + kb + half * 16);
      *(float4*)&va[4]  = *(const float4*)(vrow + kb + half * 16 + 4);
      *(float4*)&va[8]  = *(const float4*)(vrow + kb + half * 16 + 8);
      *(float4*)&va[12] = *(const float4*)(vrow + kb + half * 16 + 12);
      Frag vhi, vlo, thi, tlo;
#pragma unroll
      for (int jj = 0; jj < 8; ++jj) {      // pack two elements per u32 write
        unsigned int p0 = split_bf16(va[2 * jj]);
        unsigned int p1 = split_bf16(va[2 * jj + 1]);
        vhi.d[jj] = (p0 & 0xFFFFu) | ((p1 & 0xFFFFu) << 16);
        vlo.d[jj] = (p0 >> 16)     | (p1 & 0xFFFF0000u);
      }
      const int bi = (kbc * ROWS + l16) * 4 + half * 2;   // uint4 units
      ld_frag(thi, afH4 + bi);
      ld_frag(tlo, afL4 + bi);
      acc = WMMA_BF16(vhi.b, thi.b, acc);
      acc = WMMA_BF16(vhi.b, tlo.b, acc);
      acc = WMMA_BF16(vlo.b, thi.b, acc);
    }
#pragma unroll
    for (int r = 0; r < 8; ++r) {
      int c = cBase + r + 8 * half;
      Ob[(size_t)c * L_ + lBase + l16] = acc[r];
    }
  }
}

extern "C" void kernel_launch(void* const* d_in, const int* in_sizes, int n_in,
                              void* d_out, int out_size, void* d_ws, size_t ws_size,
                              hipStream_t stream) {
  const float* Q   = (const float*)d_in[0];
  const float* K   = (const float*)d_in[1];
  const float* V   = (const float*)d_in[2];
  const float* pad = (const float*)d_in[3];
  float* out  = (float*)d_out;                       // [B, C, L]
  float* attn = out + (size_t)B_ * C_ * L_;          // [B, L, L]

  const size_t lds_bytes = 272 * 1024;               // <= 320KB per WGP
  (void)hipFuncSetAttribute((const void*)attn_fused_kernel,
                            hipFuncAttributeMaxDynamicSharedMemorySize,
                            (int)lds_bytes);

  dim3 grid(B_ * (L_ / ROWS));   // 8 * 128 = 1024 workgroups
  dim3 block(NTHR);              // 8 wave32
  attn_fused_kernel<<<grid, block, lds_bytes, stream>>>(Q, K, V, pad, out, attn);
}
